// MarlinLayer_2190433321407
// MI455X (gfx1250) — matrix operand order repaired and checked by reading, not compiled, not run on hardware
//
#include <hip/hip_runtime.h>
#include <stdint.h>

#define M_DIM 8192
#define N_DIM 4096
#define K_DIM 4096
#define GRP   128

#define BM 128
#define BN 128
#define BK 64

// fused-fallback LDS stride (ushorts)
#define LDA 80
#define LDB 80
// async-GEMM LDS stride: 72 ushorts = 144B = 36 dwords; gcd(36,64)=4 -> 16
// consecutive rows map to 16 distinct 4-dword bank groups (conflict-free b128)
#define LDT 72

typedef __attribute__((ext_vector_type(16))) __bf16         v16bf;
typedef __attribute__((ext_vector_type(8)))  float          v8f;
typedef __attribute__((ext_vector_type(16))) unsigned short v16u;
typedef __attribute__((ext_vector_type(8)))  unsigned short u16x8;
typedef __attribute__((ext_vector_type(4)))  unsigned short u16x4;
typedef __attribute__((ext_vector_type(4)))  float          float4v;
typedef __attribute__((ext_vector_type(4)))  int            int4v;

// async-copy builtin wants v4i32 pointers in explicit address spaces
typedef __attribute__((address_space(1))) int4v* gbl_v4i;   // global (addrspace 1)
typedef __attribute__((address_space(3))) int4v* lds_v4i;   // LDS    (addrspace 3)

// f32 -> bf16 with round-to-nearest-even
static __device__ __forceinline__ unsigned short f2bf(float f) {
    uint32_t u = __builtin_bit_cast(uint32_t, f);
    u += 0x7FFFu + ((u >> 16) & 1u);
    return (unsigned short)(u >> 16);
}

// ---------------------------------------------------------------------------
// Phase 1a: A (M,K) f32 -> bf16, done ONCE (instead of once per N-tile)
// ---------------------------------------------------------------------------
__global__ __launch_bounds__(256)
void conv_a_bf16(const float* __restrict__ A, unsigned short* __restrict__ Abf) {
    const size_t idx = (size_t)blockIdx.x * 256 + threadIdx.x;   // 8-elem chunk
    const float4v a0 = *(const float4v*)(A + idx * 8);
    const float4v a1 = *(const float4v*)(A + idx * 8 + 4);
    u16x8 w;
#pragma unroll
    for (int e = 0; e < 4; ++e) { w[e] = f2bf(a0[e]); w[e + 4] = f2bf(a1[e]); }
    *(u16x8*)(Abf + idx * 8) = w;
}

// ---------------------------------------------------------------------------
// Phase 1b: B int4 (K/8,N) -> bf16 (N,K) transposed via LDS, done ONCE
// (instead of once per M-tile). K-contiguous output => GEMM can async-copy.
// ---------------------------------------------------------------------------
#define DQ_N  256
#define DQ_KP 8
__global__ __launch_bounds__(256)
void dequant_b_bf16(const int* __restrict__ Bq, const float* __restrict__ S,
                    unsigned short* __restrict__ Bt) {
    __shared__ unsigned short T[DQ_N * LDT];   // 256 x 72 ushort = 36,864 B

    const int tid = threadIdx.x;
    const int n0  = blockIdx.x * DQ_N;
    const int kp0 = blockIdx.y * DQ_KP;        // packed rows; k0 = kp0*8
    const int g   = (kp0 * 8) / GRP;           // constant over the tile
    const int kp  = tid >> 5;                  // 0..7
    const int nl0 = tid & 31;

#pragma unroll
    for (int j = 0; j < 8; ++j) {
        const int   nl = nl0 + j * 32;
        const int   q  = Bq[(size_t)(kp0 + kp) * N_DIM + n0 + nl];
        const float sc = S[(size_t)g * N_DIM + n0 + nl];
        u16x8 w;
#pragma unroll
        for (int t = 0; t < 8; ++t)
            w[t] = f2bf(((float)((q >> (4 * t)) & 0xF) - 8.0f) * sc);
        *(u16x8*)&T[nl * LDT + kp * 8] = w;
    }
    __syncthreads();

    // coalesced write-out: each row = 64 ushort = 8 x b128 chunks
#pragma unroll
    for (int p = 0; p < 8; ++p) {
        const int idx = tid + p * 256;
        const int row = idx >> 3, c = idx & 7;
        u16x8 v = *(const u16x8*)&T[row * LDT + c * 8];
        *(u16x8*)(Bt + (size_t)(n0 + row) * K_DIM + kp0 * 8 + c * 8) = v;
    }
}

// ---------------------------------------------------------------------------
// Phase 2: bf16 x bf16 GEMM; tiles streamed with global_load_async_to_lds_b128
// into double-buffered LDS; inner loop is ds_load_b128 + v_wmma only.
// ---------------------------------------------------------------------------
__global__ __launch_bounds__(256)
void gemm_bf16_wmma_async(const unsigned short* __restrict__ Abf,
                          const unsigned short* __restrict__ Bt,
                          float* __restrict__ C) {
    __shared__ unsigned short As[2][BM * LDT];   // 2 x 9,216 ushort
    __shared__ unsigned short Bs[2][BN * LDT];   // total LDS = 73,728 B

    const int tid  = threadIdx.x;
    const int lane = tid & 31;
    const int wave = tid >> 5;
    const int wm   = wave & 3;
    const int wn   = wave >> 2;
    const int m0   = blockIdx.y * BM;
    const int n0   = blockIdx.x * BN;

    // one tile = 128 rows x 64 bf16 = 1024 x 16B chunks; 4 chunks/thread/operand
    auto issue_async = [&](int kt, int buf) {
        const int k0 = kt * BK;
#pragma unroll
        for (int i = 0; i < 4; ++i) {
            const int idx = tid + i * 256;
            const int r = idx >> 3, c = idx & 7;
            __builtin_amdgcn_global_load_async_to_lds_b128(
                (gbl_v4i)(Abf + (size_t)(m0 + r) * K_DIM + k0 + c * 8),
                (lds_v4i)&As[buf][r * LDT + c * 8], 0, 0);
        }
#pragma unroll
        for (int i = 0; i < 4; ++i) {
            const int idx = tid + i * 256;
            const int r = idx >> 3, c = idx & 7;
            __builtin_amdgcn_global_load_async_to_lds_b128(
                (gbl_v4i)(Bt + (size_t)(n0 + r) * K_DIM + k0 + c * 8),
                (lds_v4i)&Bs[buf][r * LDT + c * 8], 0, 0);
        }
    };

    // ISA 16-bit A 16x32 layout: lane l -> row l%16; lanes>=16 shift K by +8;
    // chunks at {base, base+16}
    auto load_a_frag = [&](int buf, int mrow, int kk) -> v16bf {
        const unsigned short* base =
            &As[buf][(mrow + (lane & 15)) * LDT + kk + ((lane >> 4) << 3)];
        u16x8 lo = *(const u16x8*)(base);
        u16x8 hi = *(const u16x8*)(base + 16);
        v16u t;
#pragma unroll
        for (int i = 0; i < 8; ++i) { t[i] = lo[i]; t[i + 8] = hi[i]; }
        return __builtin_bit_cast(v16bf, t);
    };
    // ISA 16-bit B 32x16 layout: lane l -> col l%16; lanes>=16 shift K by +8; chunks at {base, base+16}
    auto load_b_frag = [&](int buf, int ncol, int kk) -> v16bf {
        const unsigned short* base =
            &Bs[buf][(ncol + (lane & 15)) * LDT + kk + ((lane >> 4) << 3)];
        u16x8 lo = *(const u16x8*)(base);
        u16x8 hi = *(const u16x8*)(base + 16);
        v16u t;
#pragma unroll
        for (int i = 0; i < 8; ++i) { t[i] = lo[i]; t[i + 8] = hi[i]; }
        return __builtin_bit_cast(v16bf, t);
    };

    v8f acc[2][4];
#pragma unroll
    for (int mf = 0; mf < 2; ++mf)
#pragma unroll
        for (int nf = 0; nf < 4; ++nf)
#pragma unroll
            for (int i = 0; i < 8; ++i) acc[mf][nf][i] = 0.0f;

    constexpr int NT = K_DIM / BK;   // 64

    issue_async(0, 0);
    __builtin_amdgcn_s_wait_asynccnt(0);
    __syncthreads();

    int buf = 0;
#pragma unroll 1
    for (int kt = 0; kt < NT; ++kt) {
        if (kt + 1 < NT) issue_async(kt + 1, buf ^ 1);   // overlaps with WMMAs below
        if (kt + 2 < NT) {
            __builtin_prefetch(Abf + (size_t)(m0 + (tid >> 1)) * K_DIM + (kt + 2) * BK, 0, 1);
            __builtin_prefetch(Bt + (size_t)(n0 + (tid >> 1)) * K_DIM + (kt + 2) * BK, 0, 1);
        }

#pragma unroll
        for (int kk = 0; kk < BK; kk += 32) {
            v16bf a0 = load_a_frag(buf, wm * 32 + 0,  kk);
            v16bf a1 = load_a_frag(buf, wm * 32 + 16, kk);
#pragma unroll
            for (int nf = 0; nf < 4; ++nf) {
                v16bf b = load_b_frag(buf, wn * 64 + nf * 16, kk);
                acc[0][nf] = __builtin_amdgcn_wmma_f32_16x16x32_bf16(
                    false, a0, false, b, (short)0, acc[0][nf], false, false);
                acc[1][nf] = __builtin_amdgcn_wmma_f32_16x16x32_bf16(
                    false, a1, false, b, (short)0, acc[1][nf], false, false);
            }
        }

        if (kt + 1 < NT) __builtin_amdgcn_s_wait_asynccnt(0);
        __syncthreads();
        buf ^= 1;
    }

    // C layout (16x16 f32): VGPR i: lanes 0-15 -> M=i, lanes 16-31 -> M=8+i; N=lane%16
#pragma unroll
    for (int mf = 0; mf < 2; ++mf)
#pragma unroll
        for (int nf = 0; nf < 4; ++nf) {
            const int row0 = m0 + wm * 32 + mf * 16 + ((lane >> 4) << 3);
            const int col  = n0 + wn * 64 + nf * 16 + (lane & 15);
#pragma unroll
            for (int i = 0; i < 8; ++i)
                C[(size_t)(row0 + i) * N_DIM + col] = acc[mf][nf][i];
        }
}

// ---------------------------------------------------------------------------
// Fallback: fused dequant GEMM (used only if d_ws is too small)
// ---------------------------------------------------------------------------
__global__ __launch_bounds__(256)
void marlin_w4a16_wmma_bf16(const float* __restrict__ A,
                            const int*   __restrict__ Bq,
                            const float* __restrict__ S,
                            float*       __restrict__ C) {
    __shared__ unsigned short As[BM * LDA];
    __shared__ unsigned short Bs[BN * LDB];

    const int tid  = threadIdx.x;
    const int lane = tid & 31;
    const int wave = tid >> 5;
    const int wm   = wave & 3;
    const int wn   = wave >> 2;
    const int m0   = blockIdx.y * BM;
    const int n0   = blockIdx.x * BN;

    float4v a_reg[8];
    int4v   b_reg;
    float4v s_reg;
    const int kpBase = tid >> 5;
    const int nB     = (tid & 31) * 4;

    auto g_load = [&](int kt) {
        const int k0 = kt * BK;
#pragma unroll
        for (int i = 0; i < 8; ++i) {
            int idx = tid + i * 256;
            int r = idx >> 4, cv = idx & 15;
            a_reg[i] = *(const float4v*)(A + (size_t)(m0 + r) * K_DIM + k0 + cv * 4);
        }
        b_reg = *(const int4v*)(Bq + (size_t)(k0 / 8 + kpBase) * N_DIM + n0 + nB);
        s_reg = *(const float4v*)(S + (size_t)(k0 / GRP) * N_DIM + n0 + nB);
    };

    auto s_store = [&]() {
#pragma unroll
        for (int i = 0; i < 8; ++i) {
            int idx = tid + i * 256;
            int r = idx >> 4, cv = idx & 15;
            u16x4 p;
#pragma unroll
            for (int e = 0; e < 4; ++e) p[e] = f2bf(a_reg[i][e]);
            *(u16x4*)&As[r * LDA + cv * 4] = p;
        }
#pragma unroll
        for (int j = 0; j < 4; ++j) {
            const int   q  = b_reg[j];
            const float sc = s_reg[j];
            u16x8 w;
#pragma unroll
            for (int t = 0; t < 8; ++t)
                w[t] = f2bf(((float)((q >> (4 * t)) & 0xF) - 8.0f) * sc);
            *(u16x8*)&Bs[(nB + j) * LDB + kpBase * 8] = w;
        }
    };

    auto load_a_frag = [&](int mrow, int kk) -> v16bf {
        const unsigned short* base =
            &As[(mrow + (lane & 15)) * LDA + kk + ((lane >> 4) << 3)];
        u16x8 lo = *(const u16x8*)(base);
        u16x8 hi = *(const u16x8*)(base + 16);
        v16u t;
#pragma unroll
        for (int i = 0; i < 8; ++i) { t[i] = lo[i]; t[i + 8] = hi[i]; }
        return __builtin_bit_cast(v16bf, t);
    };
    auto load_b_frag = [&](int ncol, int kk) -> v16bf {
        const unsigned short* base =
            &Bs[(ncol + (lane & 15)) * LDB + kk + ((lane >> 4) << 3)];
        u16x8 lo = *(const u16x8*)(base);
        u16x8 hi = *(const u16x8*)(base + 16);
        v16u t;
#pragma unroll
        for (int i = 0; i < 8; ++i) { t[i] = lo[i]; t[i + 8] = hi[i]; }
        return __builtin_bit_cast(v16bf, t);
    };

    v8f acc[2][4];
#pragma unroll
    for (int mf = 0; mf < 2; ++mf)
#pragma unroll
        for (int nf = 0; nf < 4; ++nf)
#pragma unroll
            for (int i = 0; i < 8; ++i) acc[mf][nf][i] = 0.0f;

    constexpr int NT = K_DIM / BK;

    g_load(0);
    s_store();
    __syncthreads();

#pragma unroll 1
    for (int kt = 0; kt < NT; ++kt) {
        if (kt + 1 < NT) g_load(kt + 1);

#pragma unroll
        for (int kk = 0; kk < BK; kk += 32) {
            v16bf a0 = load_a_frag(wm * 32 + 0,  kk);
            v16bf a1 = load_a_frag(wm * 32 + 16, kk);
#pragma unroll
            for (int nf = 0; nf < 4; ++nf) {
                v16bf b = load_b_frag(wn * 64 + nf * 16, kk);
                acc[0][nf] = __builtin_amdgcn_wmma_f32_16x16x32_bf16(
                    false, a0, false, b, (short)0, acc[0][nf], false, false);
                acc[1][nf] = __builtin_amdgcn_wmma_f32_16x16x32_bf16(
                    false, a1, false, b, (short)0, acc[1][nf], false, false);
            }
        }
        __syncthreads();
        if (kt + 1 < NT) {
            s_store();
            __syncthreads();
        }
    }

#pragma unroll
    for (int mf = 0; mf < 2; ++mf)
#pragma unroll
        for (int nf = 0; nf < 4; ++nf) {
            const int row0 = m0 + wm * 32 + mf * 16 + ((lane >> 4) << 3);
            const int col  = n0 + wn * 64 + nf * 16 + (lane & 15);
#pragma unroll
            for (int i = 0; i < 8; ++i)
                C[(size_t)(row0 + i) * N_DIM + col] = acc[mf][nf][i];
        }
}

extern "C" void kernel_launch(void* const* d_in, const int* in_sizes, int n_in,
                              void* d_out, int out_size, void* d_ws, size_t ws_size,
                              hipStream_t stream) {
    (void)in_sizes; (void)n_in; (void)out_size;
    const float* A  = (const float*)d_in[0];
    const int*   Bq = (const int*)d_in[1];
    const float* S  = (const float*)d_in[2];
    float*       C  = (float*)d_out;

    const size_t needWs =
        ((size_t)M_DIM * K_DIM + (size_t)N_DIM * K_DIM) * sizeof(unsigned short);

    if (ws_size >= needWs) {
        unsigned short* Abf = (unsigned short*)d_ws;
        unsigned short* Bt  = Abf + (size_t)M_DIM * K_DIM;

        conv_a_bf16<<<(int)(((size_t)M_DIM * K_DIM / 8) / 256), 256, 0, stream>>>(A, Abf);
        dequant_b_bf16<<<dim3(N_DIM / DQ_N, (K_DIM / 8) / DQ_KP), 256, 0, stream>>>(Bq, S, Bt);
        gemm_bf16_wmma_async<<<dim3(N_DIM / BN, M_DIM / BM), 256, 0, stream>>>(Abf, Bt, C);
    } else {
        marlin_w4a16_wmma_bf16<<<dim3(N_DIM / BN, M_DIM / BM), 256, 0, stream>>>(A, Bq, S, C);
    }
}
